// TransformerBlock_76905684402472
// MI455X (gfx1250) — matrix operand order repaired and checked by reading, not compiled, not run on hardware
//
#include <hip/hip_runtime.h>
#include <hip/hip_bf16.h>

typedef __attribute__((ext_vector_type(16))) _Float16 v16h;
typedef __attribute__((ext_vector_type(8)))  _Float16 v8h;
typedef __attribute__((ext_vector_type(8)))  float    v8f;

#define DIM       1024
#define NUM_HEADS 16
#define HEAD_DIM  64
#define MLP_DIM   4096
#define SEQ_N     8192
#define BATCH     2
#define M_TOT     (BATCH * SEQ_N)      // 16384 rows
#define LN_EPS    1e-5f
#define ATTN_SCALE 0.125f              // 64^-0.5

// ---------------------------------------------------------------------------
// Async global -> LDS copy of 16 bytes per lane (CDNA5 ASYNCcnt path).
// VDST VGPR carries the LDS byte offset (generic LDS addr bits [31:0]).
// ---------------------------------------------------------------------------
__device__ __forceinline__ void async_copy_b128(const _Float16* gsrc, _Float16* ldst) {
    unsigned       loff = (unsigned)(uintptr_t)ldst;            // LDS byte offset
    unsigned long long ga = (unsigned long long)(uintptr_t)gsrc;
    asm volatile("global_load_async_to_lds_b128 %0, %1, off"
                 :: "v"(loff), "v"(ga) : "memory");
}
__device__ __forceinline__ void wait_async_le4() {
    asm volatile("s_wait_asynccnt 0x4" ::: "memory");
}
__device__ __forceinline__ void wait_async_0() {
    asm volatile("s_wait_asynccnt 0x0" ::: "memory");
}

// ---------------------------------------------------------------------------
// Weight transpose + f32 -> f16 convert:  wt[n*K + k] = (f16) w[k*N + n]
// ---------------------------------------------------------------------------
__global__ void wtrans_kernel(const float* __restrict__ w, _Float16* __restrict__ wt,
                              int K, int Nc) {
    size_t idx = (size_t)blockIdx.x * blockDim.x + threadIdx.x;
    size_t total = (size_t)K * Nc;
    if (idx >= total) return;
    int n = (int)(idx / K);
    int k = (int)(idx % K);
    wt[idx] = (_Float16)w[(size_t)k * Nc + n];
}

// ---------------------------------------------------------------------------
// LayerNorm, one wave per row of 1024, f32 in -> f16 out
// ---------------------------------------------------------------------------
__global__ void ln_kernel(const float* __restrict__ x, const float* __restrict__ g,
                          const float* __restrict__ b, _Float16* __restrict__ out) {
    const int wave = threadIdx.x >> 5;
    const int lane = threadIdx.x & 31;
    const size_t t = (size_t)blockIdx.x * 8 + wave;
    const float* px = x + t * DIM;

    float vals[32];
    float s = 0.f, sq = 0.f;
#pragma unroll
    for (int i = 0; i < 32; ++i) {
        float v = px[i * 32 + lane];
        vals[i] = v;
        s += v;
        sq += v * v;
    }
#pragma unroll
    for (int mask = 1; mask < 32; mask <<= 1) {
        s  += __shfl_xor(s,  mask, 32);
        sq += __shfl_xor(sq, mask, 32);
    }
    const float mean = s * (1.f / DIM);
    const float var  = sq * (1.f / DIM) - mean * mean;
    const float rs   = rsqrtf(var + LN_EPS);

    _Float16* po = out + t * DIM;
#pragma unroll
    for (int i = 0; i < 32; ++i) {
        int c = i * 32 + lane;
        po[c] = (_Float16)((vals[i] - mean) * rs * g[c] + b[c]);
    }
}

// ---------------------------------------------------------------------------
// WMMA GEMM:  C[M,N] = A[M,K](f16) * BT[N,K](f16, pre-transposed) + bias
//
// Block tile 128M x 128N, K-step 32, double-buffered LDS (2x 8KB A + 2x 8KB B)
// filled via global_load_async_to_lds_b128 (4 async instr per wave per fill).
// 8 waves = 4(M) x 2(N); each wave computes 32x64 via 2 A-frags x 4 B-frags
// -> 8 v_wmma_f32_16x16x32_f16 per K-step.
//
// EPI 0: +bias            -> f16 out
// EPI 1: +bias +residual  -> f32 out
// EPI 2: +bias, GELU(erf) -> f16 out
// ---------------------------------------------------------------------------
template <int EPI>
__global__ void gemm_wmma(const _Float16* __restrict__ A,
                          const _Float16* __restrict__ BT,
                          const float* __restrict__ bias,
                          const float* __restrict__ resid,
                          float* __restrict__ outF32,
                          _Float16* __restrict__ outF16,
                          int M, int N, int K) {
    __shared__ __align__(128) _Float16 sA[2][128 * 32];   // 8KB each
    __shared__ __align__(128) _Float16 sB[2][128 * 32];

    const int tid   = threadIdx.x;
    const int lane  = tid & 31;
    const int waveI = tid >> 5;
    const int waveM = waveI & 3;     // 0..3 -> 32 rows each
    const int waveN = waveI >> 2;    // 0..1 -> 64 cols each
    const int l15   = lane & 15;
    const int g     = lane >> 4;

    const int blockRow = blockIdx.x * 128;
    const int blockCol = blockIdx.y * 128;

    // Copy plan: 1024 16B chunks per K-step (512 A + 512 B); 4 per thread.
    const int c0 = tid, c1 = tid + 256;
    const int ar0 = c0 >> 2, as0 = (c0 & 3) * 8;   // row 0..127, halfword seg
    const int ar1 = c1 >> 2, as1 = (c1 & 3) * 8;
    const _Float16* gA0 = A  + (size_t)(blockRow + ar0) * K + as0;
    const _Float16* gA1 = A  + (size_t)(blockRow + ar1) * K + as1;
    const _Float16* gB0 = BT + (size_t)(blockCol + ar0) * K + as0;
    const _Float16* gB1 = BT + (size_t)(blockCol + ar1) * K + as1;
    const int lA0 = ar0 * 32 + as0;
    const int lA1 = ar1 * 32 + as1;

    v8f acc[2][4];
#pragma unroll
    for (int t = 0; t < 2; ++t)
#pragma unroll
        for (int j = 0; j < 4; ++j) acc[t][j] = (v8f){};

    const int ksteps = K / 32;

    // prologue: fill buffer 0
    async_copy_b128(gA0, &sA[0][lA0]);
    async_copy_b128(gA1, &sA[0][lA1]);
    async_copy_b128(gB0, &sB[0][lA0]);
    async_copy_b128(gB1, &sB[0][lA1]);

    for (int i = 0; i < ksteps; ++i) {
        const int cur = i & 1;
        const int nxt = cur ^ 1;
        if (i + 1 < ksteps) {
            const int kn = (i + 1) * 32;
            async_copy_b128(gA0 + kn, &sA[nxt][lA0]);
            async_copy_b128(gA1 + kn, &sA[nxt][lA1]);
            async_copy_b128(gB0 + kn, &sB[nxt][lA0]);
            async_copy_b128(gB1 + kn, &sB[nxt][lA1]);
            wait_async_le4();          // current buffer's 4 loads retired
        } else {
            wait_async_0();
        }
        __syncthreads();               // tile visible to all waves

        // A fragments (rows waveM*32 + t*16): chunks K[16g..16g+7], K[16g+8..+7]
        const _Float16* pa0 = &sA[cur][(waveM * 32 + l15) * 32 + g * 16];
        v8h a0lo = *(const v8h*)pa0;
        v8h a0hi = *(const v8h*)(pa0 + 8);
        v16h af0 = __builtin_shufflevector(a0lo, a0hi, 0, 1, 2, 3, 4, 5, 6, 7,
                                                        8, 9, 10, 11, 12, 13, 14, 15);
        const _Float16* pa1 = pa0 + 16 * 32;
        v8h a1lo = *(const v8h*)pa1;
        v8h a1hi = *(const v8h*)(pa1 + 8);
        v16h af1 = __builtin_shufflevector(a1lo, a1hi, 0, 1, 2, 3, 4, 5, 6, 7,
                                                        8, 9, 10, 11, 12, 13, 14, 15);
#pragma unroll
        for (int j = 0; j < 4; ++j) {
            v16h bf = *(const v16h*)&sB[cur][(waveN * 64 + j * 16 + l15) * 32 + g * 16];
            acc[0][j] = __builtin_amdgcn_wmma_f32_16x16x32_f16(false, af0, false, bf,
                                                               (short)0, acc[0][j], false, false);
            acc[1][j] = __builtin_amdgcn_wmma_f32_16x16x32_f16(false, af1, false, bf,
                                                               (short)0, acc[1][j], false, false);
        }
        __syncthreads();               // reads done before buffer reuse
    }

    // Epilogue. C layout: col = l15, row = r + 8*g within each 16x16 tile.
#pragma unroll
    for (int t = 0; t < 2; ++t) {
#pragma unroll
        for (int j = 0; j < 4; ++j) {
            const int col = blockCol + waveN * 64 + j * 16 + l15;
            const float bv = bias[col];
#pragma unroll
            for (int r = 0; r < 8; ++r) {
                const int row = blockRow + waveM * 32 + t * 16 + r + 8 * g;
                float cv = acc[t][j][r] + bv;
                const size_t oi = (size_t)row * N + col;
                if (EPI == 0) {
                    outF16[oi] = (_Float16)cv;
                } else if (EPI == 1) {
                    outF32[oi] = cv + resid[oi];
                } else { // exact GELU
                    cv = 0.5f * cv * (1.f + erff(cv * 0.70710678118654752f));
                    outF16[oi] = (_Float16)cv;
                }
            }
        }
    }
}

// ---------------------------------------------------------------------------
// Per-token attention over heads: RoPE(q,k) -> 16x16 scores (WMMA) -> softmax
// -> attn @ v. One wave per token, 8 tokens per block. Output f16 in the
// reference's transpose(0,2,1,3).reshape memory order (B,H,N,D).
// ---------------------------------------------------------------------------
__global__ void attn_kernel(const _Float16* __restrict__ qkv, _Float16* __restrict__ a16) {
    __shared__ __align__(64) _Float16 sQ[8][NUM_HEADS * HEAD_DIM];
    __shared__ __align__(64) _Float16 sK[8][NUM_HEADS * HEAD_DIM];
    __shared__ __align__(64) _Float16 sV[8][NUM_HEADS * HEAD_DIM];
    __shared__ __align__(64) float    sAttn[8][NUM_HEADS * NUM_HEADS];

    const int wave = threadIdx.x >> 5;
    const int lane = threadIdx.x & 31;
    const int t = blockIdx.x * 8 + wave;
    const int b = t >> 13;
    const int n = t & (SEQ_N - 1);

    const _Float16* row = qkv + (size_t)t * (3 * DIM);
    const int h = lane >> 1, half = lane & 1;

#pragma unroll 4
    for (int j = 0; j < 32; ++j) {
        float q1 = (float)row[h * 64 + j];
        float q2 = (float)row[h * 64 + j + 32];
        float k1 = (float)row[DIM + h * 64 + j];
        float k2 = (float)row[DIM + h * 64 + j + 32];
        float invf = __expf((float)j * -0.28782313662425572f); // 10000^(-j/32)
        float sn, cs;
        __sincosf((float)n * invf, &sn, &cs);
        if (half == 0) {
            sQ[wave][h * 64 + j] = (_Float16)(q1 * cs - q2 * sn);
            sK[wave][h * 64 + j] = (_Float16)(k1 * cs - k2 * sn);
        } else {
            sQ[wave][h * 64 + j + 32] = (_Float16)(q1 * sn + q2 * cs);
            sK[wave][h * 64 + j + 32] = (_Float16)(k1 * sn + k2 * cs);
        }
        sV[wave][h * 64 + half * 32 + j] = row[2 * DIM + h * 64 + half * 32 + j];
    }
    __syncthreads();

    const int l15 = lane & 15, g = lane >> 4;
    v8f acc = {};
#pragma unroll
    for (int ks = 0; ks < 64; ks += 32) {
        const _Float16* pq = &sQ[wave][l15 * 64 + ks + g * 16];
        v8h alo = *(const v8h*)pq;
        v8h ahi = *(const v8h*)(pq + 8);
        v16h af = __builtin_shufflevector(alo, ahi, 0, 1, 2, 3, 4, 5, 6, 7,
                                                    8, 9, 10, 11, 12, 13, 14, 15);
        const _Float16* pk = &sK[wave][l15 * 64 + ks + g * 16];
        v16h bf = *(const v16h*)pk;
        acc = __builtin_amdgcn_wmma_f32_16x16x32_f16(false, af, false, bf, (short)0, acc, false, false);
    }

#pragma unroll
    for (int r = 0; r < 8; ++r) {
        float v = acc[r] * ATTN_SCALE;
        float mx = v;
#pragma unroll
        for (int mask = 1; mask < 16; mask <<= 1) mx = fmaxf(mx, __shfl_xor(mx, mask, 32));
        float e = __expf(v - mx);
        float sm = e;
#pragma unroll
        for (int mask = 1; mask < 16; mask <<= 1) sm += __shfl_xor(sm, mask, 32);
        sAttn[wave][(r + 8 * g) * 16 + l15] = e / sm;
    }
    __syncthreads();

    float od[32];
#pragma unroll
    for (int d = 0; d < 32; ++d) od[d] = 0.f;
    for (int gg = 0; gg < 16; ++gg) {
        const float aw = sAttn[wave][h * 16 + gg];
        const _Float16* pv = &sV[wave][gg * 64 + half * 32];
#pragma unroll
        for (int d = 0; d < 32; ++d) od[d] += aw * (float)pv[d];
    }
    const size_t base = (size_t)b * (NUM_HEADS * SEQ_N * HEAD_DIM)
                      + (size_t)h * (SEQ_N * HEAD_DIM)
                      + (size_t)n * HEAD_DIM + half * 32;
#pragma unroll
    for (int d = 0; d < 32; ++d) a16[base + d] = (_Float16)od[d];
}

// ---------------------------------------------------------------------------
// Host-side orchestration
// ---------------------------------------------------------------------------
static inline size_t align256(size_t x) { return (x + 255) & ~(size_t)255; }

extern "C" void kernel_launch(void* const* d_in, const int* in_sizes, int n_in,
                              void* d_out, int out_size, void* d_ws, size_t ws_size,
                              hipStream_t stream) {
    const float* x      = (const float*)d_in[0];
    const float* ln1_g  = (const float*)d_in[1];
    const float* ln1_b  = (const float*)d_in[2];
    const float* qkv_w  = (const float*)d_in[3];
    const float* qkv_b  = (const float*)d_in[4];
    const float* proj_w = (const float*)d_in[5];
    const float* proj_b = (const float*)d_in[6];
    const float* ln2_g  = (const float*)d_in[7];
    const float* ln2_b  = (const float*)d_in[8];
    const float* mlp_w1 = (const float*)d_in[9];
    const float* mlp_b1 = (const float*)d_in[10];
    const float* mlp_w2 = (const float*)d_in[11];
    const float* mlp_b2 = (const float*)d_in[12];
    float* out = (float*)d_out;

    char* ws = (char*)d_ws;
    size_t off = 0;
    _Float16* h1     = (_Float16*)(ws + off); off = align256(off + (size_t)M_TOT * DIM * 2);
    _Float16* qkvT   = (_Float16*)(ws + off); off = align256(off + (size_t)3 * DIM * DIM * 2);
    _Float16* projT  = (_Float16*)(ws + off); off = align256(off + (size_t)DIM * DIM * 2);
    _Float16* mlp1T  = (_Float16*)(ws + off); off = align256(off + (size_t)MLP_DIM * DIM * 2);
    _Float16* mlp2T  = (_Float16*)(ws + off); off = align256(off + (size_t)DIM * MLP_DIM * 2);
    _Float16* qkv16  = (_Float16*)(ws + off); off = align256(off + (size_t)M_TOT * 3 * DIM * 2);
    _Float16* a16    = (_Float16*)(ws + off); off = align256(off + (size_t)M_TOT * DIM * 2);
    float*    x1     = (float*)   (ws + off); off = align256(off + (size_t)M_TOT * DIM * 4);
    _Float16* h2     = (_Float16*)(ws + off); off = align256(off + (size_t)M_TOT * DIM * 2);
    _Float16* m16    = (_Float16*)(ws + off); off = align256(off + (size_t)M_TOT * MLP_DIM * 2);
    (void)ws_size; (void)in_sizes; (void)n_in; (void)out_size;

    // 1) weights -> f16 transposed (L2-resident afterwards)
    wtrans_kernel<<<(3 * DIM * DIM) / 256, 256, 0, stream>>>(qkv_w, qkvT, DIM, 3 * DIM);
    wtrans_kernel<<<(DIM * DIM) / 256, 256, 0, stream>>>(proj_w, projT, DIM, DIM);
    wtrans_kernel<<<(DIM * MLP_DIM) / 256, 256, 0, stream>>>(mlp_w1, mlp1T, DIM, MLP_DIM);
    wtrans_kernel<<<(MLP_DIM * DIM) / 256, 256, 0, stream>>>(mlp_w2, mlp2T, MLP_DIM, DIM);

    // 2) LN1
    ln_kernel<<<M_TOT / 8, 256, 0, stream>>>(x, ln1_g, ln1_b, h1);

    // 3) qkv GEMM
    gemm_wmma<0><<<dim3(M_TOT / 128, (3 * DIM) / 128), 256, 0, stream>>>(
        h1, qkvT, qkv_b, nullptr, nullptr, qkv16, M_TOT, 3 * DIM, DIM);

    // 4) per-token attention (RoPE + softmax over heads)
    attn_kernel<<<M_TOT / 8, 256, 0, stream>>>(qkv16, a16);

    // 5) proj GEMM + bias + residual(x) -> f32 x1
    gemm_wmma<1><<<dim3(M_TOT / 128, DIM / 128), 256, 0, stream>>>(
        a16, projT, proj_b, x, x1, nullptr, M_TOT, DIM, DIM);

    // 6) LN2
    ln_kernel<<<M_TOT / 8, 256, 0, stream>>>(x1, ln2_g, ln2_b, h2);

    // 7) MLP1 GEMM + bias + GELU -> f16
    gemm_wmma<2><<<dim3(M_TOT / 128, MLP_DIM / 128), 256, 0, stream>>>(
        h2, mlp1T, mlp_b1, nullptr, nullptr, m16, M_TOT, MLP_DIM, DIM);

    // 8) MLP2 GEMM + bias + residual(x1) -> f32 out
    gemm_wmma<1><<<dim3(M_TOT / 128, DIM / 128), 256, 0, stream>>>(
        m16, mlp2T, mlp_b2, x1, out, nullptr, M_TOT, DIM, MLP_DIM);
}
